// DenseEdgeConv_13855564497355
// MI455X (gfx1250) — compile-verified, matrix operand order checked by reading, not run on hardware
//
#include <hip/hip_runtime.h>
#include <hip/hip_bf16.h>

// ---------------------------------------------------------------------------
// DenseEdgeConv for MI455X (gfx1250, wave32, WMMA)
//   B=4, N=2048, K=32, D=64, G=32, RADIUS=0.8
// ---------------------------------------------------------------------------
#define B_ 4
#define N_ 2048
#define K_ 32
#define D_ 64
#define G_ 32
#define RADIUS2 (0.8f * 0.8f)
#define TOTAL_PTS (B_ * N_)

typedef __attribute__((ext_vector_type(16))) _Float16 v16h;
typedef __attribute__((ext_vector_type(8)))  float    v8f;

// ---------------------------------------------------------------------------
// Kernel 1: fold + transpose weights to f16 in workspace.
//   wsum[n][d] = Wf[64+d][n] + Wf[128+d][n]   (x_k path, 64x32 folded)
//   wc  [n][d] = Wf[d][n]    - Wf[128+d][n]   (x_c path -> per-point bias c1)
//   w1t/w2t/wlt = plain transposes (col-major over output n, contiguous in k)
// ---------------------------------------------------------------------------
__global__ void prep_weights_kernel(const float* __restrict__ Wf,
                                    const float* __restrict__ W1,
                                    const float* __restrict__ W2,
                                    const float* __restrict__ Wl,
                                    _Float16* __restrict__ wsum,
                                    _Float16* __restrict__ wc,
                                    _Float16* __restrict__ w1t,
                                    _Float16* __restrict__ w2t,
                                    _Float16* __restrict__ wlt) {
  int t = threadIdx.x + blockIdx.x * blockDim.x;
  int stride = blockDim.x * gridDim.x;
  for (int i = t; i < 32 * 64; i += stride) {
    int n = i >> 6, d = i & 63;
    wsum[n * 64 + d] = (_Float16)(Wf[(64 + d) * 32 + n] + Wf[(128 + d) * 32 + n]);
    wc[n * 64 + d]   = (_Float16)(Wf[d * 32 + n]        - Wf[(128 + d) * 32 + n]);
  }
  for (int i = t; i < 32 * 96; i += stride) {
    int n = i / 96, k = i % 96;
    w1t[n * 96 + k] = (_Float16)W1[k * 32 + n];
  }
  for (int i = t; i < 32 * 128; i += stride) {
    int n = i >> 7, k = i & 127;
    w2t[n * 128 + k] = (_Float16)W2[k * 32 + n];
  }
  for (int i = t; i < 32 * 160; i += stride) {
    int n = i / 160, k = i % 160;
    wlt[n * 160 + k] = (_Float16)Wl[k * 32 + n];
  }
}

// ---------------------------------------------------------------------------
// Kernel 2: ball query. One wave32 per point. Ordered collection of the K
// smallest indices with d2 < R^2 via ballot + popcount rank (matches the
// reference's sort-of-indices semantics exactly). Pad with first index.
// ---------------------------------------------------------------------------
__global__ __launch_bounds__(256) void ball_query_kernel(const float* __restrict__ pos,
                                                         int* __restrict__ idx) {
  int wave = (int)((blockIdx.x * blockDim.x + threadIdx.x) >> 5);
  int lane = threadIdx.x & 31;
  if (wave >= TOTAL_PTS) return;
  int b = wave / N_;
  int n = wave % N_;
  const float* pb = pos + (size_t)b * N_ * 3;
  float px = pb[n * 3 + 0], py = pb[n * 3 + 1], pz = pb[n * 3 + 2];
  float sqn = px * px + py * py + pz * pz;

  int* row = idx + (size_t)wave * K_;
  int cnt = 0;
  int first = -1;
  for (int base = 0; base < N_ && cnt < K_; base += 32) {
    int m = base + lane;
    float qx = pb[m * 3 + 0], qy = pb[m * 3 + 1], qz = pb[m * 3 + 2];
    float sqm = qx * qx + qy * qy + qz * qz;
    float dot = px * qx + py * qy + pz * qz;
    float d2 = sqn + sqm - 2.0f * dot;
    bool within = d2 < RADIUS2;
    unsigned mask = __builtin_amdgcn_ballot_w32(within);
    if (first < 0 && mask) first = base + __builtin_ctz(mask);
    int rank = __builtin_popcount(mask & ((1u << lane) - 1u));
    if (within && (cnt + rank) < K_) row[cnt + rank] = m;
    cnt += __builtin_popcount(mask);
  }
  if (first < 0) first = n;  // cannot happen (self always within), safety
  cnt = min(cnt, K_);
  for (int t = cnt + lane; t < K_; t += 32) row[t] = first;
}

// ---------------------------------------------------------------------------
// WMMA tile loaders: CDNA5 16-bit A 16x32 layout (ISA 7.12.2):
//   lane in [0,15]: M = lane; lane in [16,31]: same M, K offset +8
//   VGPR v holds K pair: k = 2v + (v>=4 ? 8 : 0) + half*8
// B (32x16) uses the symmetric layout with N = lane&15.
// ---------------------------------------------------------------------------
__device__ __forceinline__ v16h load_A_lds(const _Float16* __restrict__ Y, int stride,
                                           int c0, int rt, int kt, int l15, int half) {
  const _Float16* p = Y + (rt * 16 + l15) * stride + c0 + kt * 32 + half * 8;
  v16h a;
#pragma unroll
  for (int v = 0; v < 8; ++v) {
    int k = 2 * v + (v >= 4 ? 8 : 0);
    a[2 * v]     = p[k];
    a[2 * v + 1] = p[k + 1];
  }
  return a;
}

__device__ __forceinline__ v16h load_B_glb(const _Float16* __restrict__ wt, int Kdim,
                                           int ct, int kt, int l15, int half) {
  const _Float16* p = wt + (ct * 16 + l15) * Kdim + kt * 32 + half * 8;
  v16h b;
#pragma unroll
  for (int v = 0; v < 8; ++v) {
    int k = 2 * v + (v >= 4 ? 8 : 0);
    b[2 * v]     = p[k];
    b[2 * v + 1] = p[k + 1];
  }
  return b;
}

// ---------------------------------------------------------------------------
// One 32x32 GEMM layer on a wave: C(32x32) = A(32xKdim) * B(Kdim x 32) + bias
// (bias distributed one-per-lane), optional relu, optional f16 store into the
// concat buffer, and K-axis (rows) max-reduction straight into global out.
// C/D layout (ISA): VGPR v -> M = rt*16 + half*8 + v, N = ct*16 + (lane&15).
// ---------------------------------------------------------------------------
template <bool RELU, bool STORE>
__device__ __forceinline__ void run_layer(const _Float16* __restrict__ Yin, int strideIn,
                                          int c0, int nkt,
                                          const _Float16* __restrict__ wt, int Kdim,
                                          float biasLane,
                                          _Float16* __restrict__ Yout, int strideOut,
                                          int coutBase,
                                          float* __restrict__ outp, int lane) {
  const int half = lane >> 4;
  const int l15 = lane & 15;
  const float bia0 = __shfl(biasLane, l15, 32);        // bias[N], N = l15
  const float bia1 = __shfl(biasLane, 16 + l15, 32);   // bias[16+N]
  v8f c00, c01, c10, c11;
#pragma unroll
  for (int v = 0; v < 8; ++v) { c00[v] = bia0; c10[v] = bia0; c01[v] = bia1; c11[v] = bia1; }

  for (int kt = 0; kt < nkt; ++kt) {
    v16h a0 = load_A_lds(Yin, strideIn, c0, 0, kt, l15, half);
    v16h a1 = load_A_lds(Yin, strideIn, c0, 1, kt, l15, half);
    v16h b0 = load_B_glb(wt, Kdim, 0, kt, l15, half);
    v16h b1 = load_B_glb(wt, Kdim, 1, kt, l15, half);
    c00 = __builtin_amdgcn_wmma_f32_16x16x32_f16(false, a0, false, b0, (short)0, c00, false, false);
    c01 = __builtin_amdgcn_wmma_f32_16x16x32_f16(false, a0, false, b1, (short)0, c01, false, false);
    c10 = __builtin_amdgcn_wmma_f32_16x16x32_f16(false, a1, false, b0, (short)0, c10, false, false);
    c11 = __builtin_amdgcn_wmma_f32_16x16x32_f16(false, a1, false, b1, (short)0, c11, false, false);
  }
  if (RELU) {
#pragma unroll
    for (int v = 0; v < 8; ++v) {
      c00[v] = fmaxf(c00[v], 0.0f); c01[v] = fmaxf(c01[v], 0.0f);
      c10[v] = fmaxf(c10[v], 0.0f); c11[v] = fmaxf(c11[v], 0.0f);
    }
  }
  if (STORE) {
#pragma unroll
    for (int v = 0; v < 8; ++v) {
      int m0 = half * 8 + v;
      int m1 = 16 + half * 8 + v;
      Yout[m0 * strideOut + coutBase + l15]      = (_Float16)c00[v];
      Yout[m0 * strideOut + coutBase + 16 + l15] = (_Float16)c01[v];
      Yout[m1 * strideOut + coutBase + l15]      = (_Float16)c10[v];
      Yout[m1 * strideOut + coutBase + 16 + l15] = (_Float16)c11[v];
    }
  }
  // max over all 32 rows (K neighbors): 8 VGPRs x 2 row-tiles, then half swap
  float m0 = c00[0], m1 = c01[0];
#pragma unroll
  for (int v = 1; v < 8; ++v) { m0 = fmaxf(m0, c00[v]); m1 = fmaxf(m1, c01[v]); }
#pragma unroll
  for (int v = 0; v < 8; ++v) { m0 = fmaxf(m0, c10[v]); m1 = fmaxf(m1, c11[v]); }
  m0 = fmaxf(m0, __shfl_xor(m0, 16, 32));
  m1 = fmaxf(m1, __shfl_xor(m1, 16, 32));
  if (lane < 16) {
    outp[l15] = m0;
    outp[16 + l15] = m1;
  }
}

// ---------------------------------------------------------------------------
// Kernel 3: main EdgeConv. One wave32 per point, 4 waves / 128-thread block.
// LDS per wave: XK[32][66] f16 (gathered neighbor feats, padded stride),
//               Y [32][162] f16 (concat buffer: [H3|H2|H1|x_c], padded),
//               xc[64] f32.
// Output layout per point: [maxH4 | maxH3 | maxH2 | maxH1 | x_c] = 192 f32.
// ---------------------------------------------------------------------------
#define WPB 4
#define XK_STRIDE 66
#define Y_STRIDE 162
#define LDS_BYTES (WPB * (32 * XK_STRIDE * 2 + 32 * Y_STRIDE * 2 + 64 * 4))

__global__ __launch_bounds__(WPB * 32) void edgeconv_kernel(
    const float* __restrict__ x, const int* __restrict__ idx,
    const _Float16* __restrict__ wsum, const _Float16* __restrict__ wc,
    const _Float16* __restrict__ w1t, const _Float16* __restrict__ w2t,
    const _Float16* __restrict__ wlt,
    const float* __restrict__ bfirst, const float* __restrict__ bias1,
    const float* __restrict__ bias2, const float* __restrict__ biasl,
    float* __restrict__ out) {
  extern __shared__ char smem[];
  const int lane = threadIdx.x & 31;
  const int w = threadIdx.x >> 5;
  const int point = blockIdx.x * WPB + w;
  if (point >= TOTAL_PTS) return;  // wave-uniform; EXEC stays all-ones for WMMA
  const int b = point / N_;
  const int n = point % N_;

  _Float16* XKbase = (_Float16*)smem;
  _Float16* Ybase  = XKbase + WPB * 32 * XK_STRIDE;
  float*    xcbase = (float*)(Ybase + WPB * 32 * Y_STRIDE);
  _Float16* XK = XKbase + w * 32 * XK_STRIDE;
  _Float16* Y  = Ybase + w * 32 * Y_STRIDE;
  float*    xc = xcbase + w * 64;

  // --- center features -> xc (f32), and broadcast into Y cols 96..159 ------
  const float* xrow = x + ((size_t)b * N_ + n) * D_;
  {
    float2 cp = ((const float2*)xrow)[lane];  // d = 2*lane, 2*lane+1
    xc[2 * lane]     = cp.x;
    xc[2 * lane + 1] = cp.y;
  }
  // --- gather neighbor features: lane handles neighbor k = lane ------------
  {
    int nb = idx[(size_t)point * K_ + lane];
    const float* xn = x + ((size_t)b * N_ + nb) * D_;
    _Float16* dst = XK + lane * XK_STRIDE;
#pragma unroll
    for (int d = 0; d < D_; d += 4) {
      float4 vv = *(const float4*)(xn + d);
      dst[d + 0] = (_Float16)vv.x;
      dst[d + 1] = (_Float16)vv.y;
      dst[d + 2] = (_Float16)vv.z;
      dst[d + 3] = (_Float16)vv.w;
    }
  }
  // Y x_c block: row 'lane' gets the shared center vector
  {
    _Float16* dst = Y + lane * Y_STRIDE + 96;
#pragma unroll
    for (int d = 0; d < D_; d += 2) {
      dst[d]     = (_Float16)xc[d];
      dst[d + 1] = (_Float16)xc[d + 1];
    }
  }
  // --- per-point folded bias c1[n] = b_first[n] + x_c . wc[:,n] ------------
  float cval;
  {
    float s = bfirst[lane];
    const _Float16* wrow = wc + lane * 64;
#pragma unroll 8
    for (int d = 0; d < D_; ++d) s += xc[d] * (float)wrow[d];
    cval = s;
  }

  float* outp = out + (size_t)point * 192;

  // L1: H1 = relu(XK @ wsum + c1)           -> Y[64:96], out[96:128]
  run_layer<true, true>(XK, XK_STRIDE, 0, 2, wsum, 64, cval,
                        Y, Y_STRIDE, 64, outp + 96, lane);
  // L2: H2 = relu(Y[64:160] @ W1 + b1)      -> Y[32:64], out[64:96]
  run_layer<true, true>(Y, Y_STRIDE, 64, 3, w1t, 96, bias1[lane],
                        Y, Y_STRIDE, 32, outp + 64, lane);
  // L3: H3 = relu(Y[32:160] @ W2 + b2)      -> Y[0:32],  out[32:64]
  run_layer<true, true>(Y, Y_STRIDE, 32, 4, w2t, 128, bias2[lane],
                        Y, Y_STRIDE, 0, outp + 32, lane);
  // L4: H4 = Y[0:160] @ W_last + b_last     -> out[0:32] (no relu, no store)
  run_layer<false, false>(Y, Y_STRIDE, 0, 5, wlt, 160, biasl[lane],
                          nullptr, 0, 0, outp + 0, lane);

  // x_c block of the output (constant over K, so max == value)
  outp[128 + lane]      = xc[lane];
  outp[128 + 32 + lane] = xc[32 + lane];
}

// ---------------------------------------------------------------------------
extern "C" void kernel_launch(void* const* d_in, const int* in_sizes, int n_in,
                              void* d_out, int out_size, void* d_ws, size_t ws_size,
                              hipStream_t stream) {
  const float* x      = (const float*)d_in[0];
  const float* pos    = (const float*)d_in[1];
  const float* Wf     = (const float*)d_in[2];
  const float* bfirst = (const float*)d_in[3];
  const float* W1     = (const float*)d_in[4];
  const float* bias1  = (const float*)d_in[5];
  const float* W2     = (const float*)d_in[6];
  const float* bias2  = (const float*)d_in[7];
  const float* Wl     = (const float*)d_in[8];
  const float* biasl  = (const float*)d_in[9];
  float* out = (float*)d_out;

  char* ws = (char*)d_ws;
  int* idx = (int*)ws;
  size_t idxBytes = (size_t)TOTAL_PTS * K_ * sizeof(int);
  _Float16* wsum = (_Float16*)(ws + idxBytes);
  _Float16* wc   = wsum + 32 * 64;
  _Float16* w1t  = wc + 32 * 64;
  _Float16* w2t  = w1t + 32 * 96;
  _Float16* wlt  = w2t + 32 * 128;

  prep_weights_kernel<<<20, 256, 0, stream>>>(Wf, W1, W2, Wl, wsum, wc, w1t, w2t, wlt);

  int bq_blocks = (TOTAL_PTS * 32 + 255) / 256;
  ball_query_kernel<<<bq_blocks, 256, 0, stream>>>(pos, idx);

  int ec_blocks = (TOTAL_PTS + WPB - 1) / WPB;
  edgeconv_kernel<<<ec_blocks, WPB * 32, LDS_BYTES, stream>>>(
      x, idx, wsum, wc, w1t, w2t, wlt, bfirst, bias1, bias2, biasl, out);
}
